// LSTM_16020228014817
// MI455X (gfx1250) — matrix operand order repaired and checked by reading, not compiled, not run on hardware
//
#include <hip/hip_runtime.h>
#include <hip/hip_bf16.h>

// ---------------------------------------------------------------------------
// LSTM forward, B=64 T=512 E=512 H=1024, bf16 WMMA (v_wmma_f32_16x16x32_bf16)
// ---------------------------------------------------------------------------

typedef __bf16 bf16_t;
typedef __attribute__((ext_vector_type(16))) __bf16 v16bf;
typedef __attribute__((ext_vector_type(8)))  __bf16 v8bf;
typedef __attribute__((ext_vector_type(8)))  float  v8f;

#define LSTM_B 64
#define LSTM_T 512
#define LSTM_E 512
#define LSTM_H 1024
#define LSTM_D 1536   // E + H
#define NCLS   10

// Workspace layout (bytes, all 32B aligned)
#define WBF_BYTES  ((size_t)4 * LSTM_H * LSTM_D * 2)   // 12,582,912
#define HBF_OFF    (WBF_BYTES)                          // bf16 h state [64][1024]
#define HBF_BYTES  ((size_t)LSTM_B * LSTM_H * 2)
#define HF_OFF     (HBF_OFF + HBF_BYTES)                // f32 h state  [64][1024]
#define HF_BYTES   ((size_t)LSTM_B * LSTM_H * 4)
#define C_OFF      (HF_OFF + HF_BYTES)                  // f32 c state  [64][1024]

__device__ __forceinline__ float sigmoidf_fast(float x) {
    return 1.0f / (1.0f + __expf(-x));
}
__device__ __forceinline__ float tanhf_fast(float x) {
    return 1.0f - 2.0f / (__expf(2.0f * x) + 1.0f);
}

// --- convert the 4 gate weight matrices fp32 -> bf16, layout [gate][H][D] ---
__global__ void lstm_convert_w(const float* __restrict__ Wf,
                               const float* __restrict__ Wi,
                               const float* __restrict__ Wc,
                               const float* __restrict__ Wo,
                               bf16_t* __restrict__ Wbf) {
    const int N = LSTM_H * LSTM_D;
    for (int i = blockIdx.x * blockDim.x + threadIdx.x; i < N;
         i += gridDim.x * blockDim.x) {
        Wbf[0 * (size_t)N + i] = (__bf16)Wf[i];
        Wbf[1 * (size_t)N + i] = (__bf16)Wi[i];
        Wbf[2 * (size_t)N + i] = (__bf16)Wc[i];
        Wbf[3 * (size_t)N + i] = (__bf16)Wo[i];
    }
}

// --- zero-initialize h (bf16 + f32) and c (f32) ---
__global__ void lstm_init_state(bf16_t* __restrict__ hbf,
                                float* __restrict__ hf,
                                float* __restrict__ cst) {
    int i = blockIdx.x * blockDim.x + threadIdx.x;
    if (i < LSTM_B * LSTM_H) {
        hbf[i] = (__bf16)0.0f;
        hf[i]  = 0.0f;
        cst[i] = 0.0f;
    }
}

// ---------------------------------------------------------------------------
// One recurrent step. Grid: 16 WGs x 256 threads. WG owns gate-output columns
// [64*blockIdx.x, +64) of all four gates. Wave w (0..7): gate = w>>1, column
// half = (w&1)*32, all 64 batch rows -> 4x2 = 8 WMMA acc tiles (64 f32 VGPRs).
// K-loop split at the concat boundary (E=512): loop 1 reads fp32 x_t and
// converts in-register; loop 2 reads the bf16 recurrent state. No branches
// inside either loop.
// ---------------------------------------------------------------------------
__global__ void __launch_bounds__(256)
lstm_step(const float* __restrict__ x,      // [B][T][E] fp32
          int t,
          const bf16_t* __restrict__ Wbf,   // [4][H][D] bf16
          const float* __restrict__ b_f,
          const float* __restrict__ b_i,
          const float* __restrict__ b_c,
          const float* __restrict__ b_o,
          bf16_t* __restrict__ hbf,         // [B][H] bf16 (in/out)
          float* __restrict__ hf,           // [B][H] f32  (out)
          float* __restrict__ cst)          // [B][H] f32  (in/out)
{
    __shared__ float gbuf[4][64][68];       // gate pre-activations (padded)

    const int tid   = threadIdx.x;
    const int wave  = tid >> 5;             // 0..7
    const int lane  = tid & 31;
    const int gate  = wave >> 1;            // 0..3  (f, i, c, o)
    const int nhalf = wave & 1;             // 0..1
    const int n0    = blockIdx.x * 64;      // column base within H
    const int lm    = lane & 15;
    const int lh    = lane >> 4;            // 0/1

    const float* bias = (gate == 0) ? b_f : (gate == 1) ? b_i
                        : (gate == 2) ? b_c : b_o;

    // accumulators: acc[mt][nt] covers rows mt*16.., cols colbase+nt*16..
    v8f acc[4][2];
    const int colbase = n0 + nhalf * 32;
#pragma unroll
    for (int nt = 0; nt < 2; ++nt) {
        float bv = bias[colbase + nt * 16 + lm];
        v8f cinit;
#pragma unroll
        for (int e = 0; e < 8; ++e) cinit[e] = bv;
#pragma unroll
        for (int mt = 0; mt < 4; ++mt) acc[mt][nt] = cinit;
    }

    const bf16_t* wbase = Wbf + (size_t)gate * LSTM_H * LSTM_D;
    // per-lane base for B fragments: col = colbase + nt*16 + lm, k = k0 + lh*16
    const bf16_t* wlane0 = wbase + (size_t)(colbase + lm) * LSTM_D + lh * 16;
    const bf16_t* wlane1 = wbase + (size_t)(colbase + 16 + lm) * LSTM_D + lh * 16;
    // per-lane base for x-region A fragments: row = mt*16+lm (batch index)
    const float* xlane = x + (size_t)lm * LSTM_T * LSTM_E
                           + (size_t)t * LSTM_E + lh * 16;
    // per-lane base for h-region A fragments
    const bf16_t* hlane = hbf + (size_t)lm * LSTM_H + lh * 16;

    // ---------------- K loop part 1: x region (K = 0 .. E) ----------------
    for (int k0 = 0; k0 < LSTM_E; k0 += 32) {
        v16bf afrag[4];
#pragma unroll
        for (int mt = 0; mt < 4; ++mt) {
            const float* xb = xlane + (size_t)mt * 16 * LSTM_T * LSTM_E + k0;
            v8f c0 = *reinterpret_cast<const v8f*>(xb);
            v8f c1 = *reinterpret_cast<const v8f*>(xb + 8);
            v16bf av;
#pragma unroll
            for (int e = 0; e < 8; ++e) {
                av[e]     = (__bf16)c0[e];
                av[e + 8] = (__bf16)c1[e];
            }
            afrag[mt] = av;
        }
        v16bf bfrag[2];
        bfrag[0] = *reinterpret_cast<const v16bf*>(wlane0 + k0);
        bfrag[1] = *reinterpret_cast<const v16bf*>(wlane1 + k0);
#pragma unroll
        for (int mt = 0; mt < 4; ++mt)
#pragma unroll
            for (int nt = 0; nt < 2; ++nt)
                acc[mt][nt] = __builtin_amdgcn_wmma_f32_16x16x32_bf16(
                    false, afrag[mt], false, bfrag[nt],
                    (short)0, acc[mt][nt],
                    false, false);
    }

    // ---------------- K loop part 2: h region (K = E .. D) ----------------
#pragma unroll 2
    for (int k0 = LSTM_E; k0 < LSTM_D; k0 += 32) {
        const int kh = k0 - LSTM_E;
        v16bf afrag[4];
#pragma unroll
        for (int mt = 0; mt < 4; ++mt) {
            const bf16_t* hb = hlane + (size_t)mt * 16 * LSTM_H + kh;
            v8bf c0 = *reinterpret_cast<const v8bf*>(hb);
            v8bf c1 = *reinterpret_cast<const v8bf*>(hb + 8);
            v16bf av;
#pragma unroll
            for (int e = 0; e < 8; ++e) {
                av[e]     = c0[e];
                av[e + 8] = c1[e];
            }
            afrag[mt] = av;
        }
        v16bf bfrag[2];
        bfrag[0] = *reinterpret_cast<const v16bf*>(wlane0 + k0);
        bfrag[1] = *reinterpret_cast<const v16bf*>(wlane1 + k0);
#pragma unroll
        for (int mt = 0; mt < 4; ++mt)
#pragma unroll
            for (int nt = 0; nt < 2; ++nt)
                acc[mt][nt] = __builtin_amdgcn_wmma_f32_16x16x32_bf16(
                    false, afrag[mt], false, bfrag[nt],
                    (short)0, acc[mt][nt],
                    false, false);
    }

    // ----- spill gate pre-activations to LDS for cross-wave combine -----
#pragma unroll
    for (int mt = 0; mt < 4; ++mt)
#pragma unroll
        for (int nt = 0; nt < 2; ++nt) {
            v8f a = acc[mt][nt];
#pragma unroll
            for (int v = 0; v < 8; ++v) {
                int m = mt * 16 + v + lh * 8;
                int n = nhalf * 32 + nt * 16 + lm;
                gbuf[gate][m][n] = a[v];
            }
        }
    __syncthreads();

    // ----- elementwise LSTM update for this WG's 64x64 slice -----
    for (int e = tid; e < 64 * 64; e += 256) {
        int m = e >> 6;
        int n = e & 63;
        float fg = sigmoidf_fast(gbuf[0][m][n]);
        float ig = sigmoidf_fast(gbuf[1][m][n]);
        float cg = tanhf_fast  (gbuf[2][m][n]);
        float og = sigmoidf_fast(gbuf[3][m][n]);
        size_t idx = (size_t)m * LSTM_H + (n0 + n);
        float cn = fg * cst[idx] + ig * cg;
        float hn = og * tanhf_fast(cn);
        cst[idx] = cn;
        hf[idx]  = hn;
        hbf[idx] = (__bf16)hn;
    }
}

// --- final FC: out[b][cls] = h[b] . W_fc[cls] + b_fc[cls] ---
__global__ void lstm_fc(const float* __restrict__ hf,
                        const float* __restrict__ Wfc,
                        const float* __restrict__ bfc,
                        float* __restrict__ out) {
    int tid = blockIdx.x * blockDim.x + threadIdx.x;
    if (tid >= LSTM_B * NCLS) return;
    int b = tid / NCLS, cls = tid % NCLS;
    const float* hr = hf + (size_t)b * LSTM_H;
    const float* wr = Wfc + (size_t)cls * LSTM_H;
    float s = bfc[cls];
    for (int k = 0; k < LSTM_H; ++k) s += hr[k] * wr[k];
    out[tid] = s;
}

extern "C" void kernel_launch(void* const* d_in, const int* in_sizes, int n_in,
                              void* d_out, int out_size, void* d_ws, size_t ws_size,
                              hipStream_t stream) {
    const float* x   = (const float*)d_in[0];
    const float* W_f = (const float*)d_in[1];
    const float* b_f = (const float*)d_in[2];
    const float* W_i = (const float*)d_in[3];
    const float* b_i = (const float*)d_in[4];
    const float* W_c = (const float*)d_in[5];
    const float* b_c = (const float*)d_in[6];
    const float* W_o = (const float*)d_in[7];
    const float* b_o = (const float*)d_in[8];
    const float* Wfc = (const float*)d_in[9];
    const float* bfc = (const float*)d_in[10];
    float* out = (float*)d_out;

    char* ws = (char*)d_ws;
    bf16_t* Wbf = (bf16_t*)(ws);
    bf16_t* hbf = (bf16_t*)(ws + HBF_OFF);
    float*  hf  = (float*)(ws + HF_OFF);
    float*  cst = (float*)(ws + C_OFF);

    // one-time (per call) prep: weight conversion + state init
    lstm_convert_w<<<1024, 256, 0, stream>>>(W_f, W_i, W_c, W_o, Wbf);
    lstm_init_state<<<(LSTM_B * LSTM_H + 255) / 256, 256, 0, stream>>>(hbf, hf, cst);

    // sequential recurrence: 16 WGs cover N=1024 gate-output columns x 4 gates
    for (int t = 0; t < LSTM_T; ++t) {
        lstm_step<<<16, 256, 0, stream>>>(x, t, Wbf, b_f, b_i, b_c, b_o,
                                          hbf, hf, cst);
    }

    lstm_fc<<<(LSTM_B * NCLS + 255) / 256, 256, 0, stream>>>(hf, Wfc, bfc, out);
}
